// CausalSelfAttention_14396730376998
// MI455X (gfx1250) — compile-verified, matrix operand order checked by reading, not run on hardware
//
#include <hip/hip_runtime.h>
#include <hip/hip_bf16.h>

typedef unsigned short u16;
typedef unsigned int   u32;

typedef __attribute__((ext_vector_type(16))) __bf16 v16bf;
typedef __attribute__((ext_vector_type(8)))  float  v8f;

// ---------------------------------------------------------------------------
// helpers
// ---------------------------------------------------------------------------
__device__ __forceinline__ u16 f2bf(float f) {
    u32 u = __float_as_uint(f);
    u += 0x7FFFu + ((u >> 16) & 1u);   // round-to-nearest-even
    return (u16)(u >> 16);
}

union FragU { v16bf v; uint4 q[2]; };

__device__ __forceinline__ v16bf load_frag(const u16* p) {
    FragU u;
    u.q[0] = *reinterpret_cast<const uint4*>(p);
    u.q[1] = *reinterpret_cast<const uint4*>(p + 8);
    return u.v;
}

__device__ __forceinline__ v8f wmma_bf16(v16bf a, v16bf b, v8f c) {
    // v_wmma_f32_16x16x32_bf16: D = A(16x32) * B(32x16) + C(16x16 f32)
    return __builtin_amdgcn_wmma_f32_16x16x32_bf16(
        /*neg_a=*/false, a, /*neg_b=*/false, b,
        /*c_mod=*/(short)0, c, /*reuse_a=*/false, /*reuse_b=*/false);
}

// 16-lane butterfly reductions via ds_swizzle (immediate XOR patterns:
// offset[14:10]=xor_mask, [9:5]=or_mask=0, [4:0]=and_mask=0x1F)
#define SWZ_XOR(x, imm) \
    __int_as_float(__builtin_amdgcn_ds_swizzle(__float_as_int(x), (imm)))

__device__ __forceinline__ float rmax16(float x) {
    x = fmaxf(x, SWZ_XOR(x, 0x041F));  // xor 1
    x = fmaxf(x, SWZ_XOR(x, 0x081F));  // xor 2
    x = fmaxf(x, SWZ_XOR(x, 0x101F));  // xor 4
    x = fmaxf(x, SWZ_XOR(x, 0x201F));  // xor 8
    return x;
}
__device__ __forceinline__ float rsum16(float x) {
    x += SWZ_XOR(x, 0x041F);
    x += SWZ_XOR(x, 0x081F);
    x += SWZ_XOR(x, 0x101F);
    x += SWZ_XOR(x, 0x201F);
    return x;
}

// ---------------------------------------------------------------------------
// fp32 -> bf16 conversion
// ---------------------------------------------------------------------------
__global__ __launch_bounds__(256) void f32_to_bf16_kernel(
    const float* __restrict__ in, u16* __restrict__ out, int n) {
    int i = blockIdx.x * blockDim.x + threadIdx.x;
    int stride = gridDim.x * blockDim.x;
    for (; i < n; i += stride) out[i] = f2bf(in[i]);
}

// ---------------------------------------------------------------------------
// GEMM: out[M,N] = A_bf16[M,K] * W_bf16[N,K]^T + bias[N]
// block = 128 threads = 4 waves; wave tile = 32(M) x 64(N); block tile 128x64.
// ---------------------------------------------------------------------------
template <bool OUT_BF16>
__global__ __launch_bounds__(128) void gemm_bf16_wmma(
    const u16* __restrict__ A, const u16* __restrict__ W,
    const float* __restrict__ bias, void* __restrict__ outp,
    int M, int N, int K) {
    const int w    = threadIdx.x >> 5;
    const int lane = threadIdx.x & 31;
    const int half = lane >> 4;
    const int r    = lane & 15;
    const int row0 = blockIdx.x * 128 + w * 32;
    const int col0 = blockIdx.y * 64;

    v8f acc[2][4];
#pragma unroll
    for (int i = 0; i < 2; ++i)
#pragma unroll
        for (int j = 0; j < 4; ++j) acc[i][j] = {};

    for (int k = 0; k < K; k += 32) {
        v16bf af[2];
#pragma unroll
        for (int i = 0; i < 2; ++i)
            af[i] = load_frag(A + (size_t)(row0 + i * 16 + r) * K + k + half * 16);
        v16bf bf[4];
#pragma unroll
        for (int j = 0; j < 4; ++j)
            bf[j] = load_frag(W + (size_t)(col0 + j * 16 + r) * K + k + half * 16);
#pragma unroll
        for (int i = 0; i < 2; ++i)
#pragma unroll
            for (int j = 0; j < 4; ++j)
                acc[i][j] = wmma_bf16(af[i], bf[j], acc[i][j]);
    }

    // epilogue: slot v on lane L holds row (i*16 + half*8 + v), col (j*16 + r)
#pragma unroll
    for (int i = 0; i < 2; ++i) {
#pragma unroll
        for (int j = 0; j < 4; ++j) {
            const int row = row0 + i * 16 + half * 8;
            const int col = col0 + j * 16 + r;
            const float bj = bias[col];
#pragma unroll
            for (int v = 0; v < 8; ++v) {
                const float val = acc[i][j][v] + bj;
                if (OUT_BF16)
                    ((u16*)outp)[(size_t)(row + v) * N + col] = f2bf(val);
                else
                    ((float*)outp)[(size_t)(row + v) * N + col] = val;
            }
        }
    }
}

// ---------------------------------------------------------------------------
// Flash attention (bf16 WMMA, fp32 softmax state).
// grid = (T/128, H, B); block = 256 = 8 waves; wave owns 16 query rows.
// qkv layout (from reference reshape): row stride 3*C=3072,
//   q at h*192+0..63, k at h*192+64..127, v at h*192+128..191.
// K tile is staged with async global->LDS DMA (ASYNCcnt); V tile is staged
// transposed through VGPRs.
// ---------------------------------------------------------------------------
__global__ __launch_bounds__(256) void attn_flash_wmma(
    const u16* __restrict__ qkv, u16* __restrict__ ybf) {
    constexpr int T = 512, C3 = 3072;
    constexpr int KS_STRIDE = 80;  // bf16 elems; 160B rows -> 32B-aligned frags
    constexpr int VT_STRIDE = 48;  // 96B rows   -> 32B-aligned frags

    __shared__ u16 Ks[32 * KS_STRIDE];       // K tile,  [key][d]
    __shared__ u16 Vt[64 * VT_STRIDE];       // V tile transposed, [d][key]
    __shared__ u16 Pb[8 * 16 * 32];          // per-wave P tile

    const int qb  = blockIdx.x;              // query block (128 rows)
    const int h   = blockIdx.y;
    const int b   = blockIdx.z;
    const int tid = threadIdx.x;
    const int w    = tid >> 5;
    const int lane = tid & 31;
    const int half = lane >> 4;
    const int r    = lane & 15;
    const int q0w  = qb * 128 + w * 16;      // wave's first query row

    // Q fragments (K-dim = Dh = 64 -> two chunks of 32)
    v16bf qf[2];
    {
        const size_t qrow = (size_t)(b * T + q0w + r) * C3 + h * 192;
        qf[0] = load_frag(qkv + qrow + 0  + half * 16);
        qf[1] = load_frag(qkv + qrow + 32 + half * 16);
    }

    float m[8], l[8];
#pragma unroll
    for (int v = 0; v < 8; ++v) { m[v] = -1e30f; l[v] = 0.0f; }
    v8f yacc[4];
#pragma unroll
    for (int j = 0; j < 4; ++j) yacc[j] = {};

    u16* Pw = Pb + w * (16 * 32);
    const int nkt = qb * 4 + 4;              // causal key-tile bound for block

    // per-thread staging coordinates: 32 keys x 64 d, 8 bf16 (16B) each
    const int skey = tid >> 3;               // 0..31
    const int sd0  = (tid & 7) * 8;          // 0..56
    const u32 ksLds = (u32)(unsigned long long)(&Ks[skey * KS_STRIDE + sd0]);

    for (int kt = 0; kt < nkt; ++kt) {
        const int k0 = kt * 32;

        __syncthreads();                     // WAR: previous tile fully consumed
        {
            const size_t src = (size_t)(b * T + k0 + skey) * C3 + h * 192;
            // --- K tile: async DMA global -> LDS (no VGPR round trip) ---
            const u32 goff = (u32)((src + 64 + sd0) * sizeof(u16));
            asm volatile("global_load_async_to_lds_b128 %0, %1, %2"
                         :: "v"(ksLds), "v"(goff), "s"(qkv)
                         : "memory");
            // --- V tile: load and transpose through VGPRs ---
            uint4 vq = *reinterpret_cast<const uint4*>(qkv + src + 128 + sd0);
            const u16* ve = reinterpret_cast<const u16*>(&vq);
#pragma unroll
            for (int j = 0; j < 8; ++j) Vt[(sd0 + j) * VT_STRIDE + skey] = ve[j];
            asm volatile("s_wait_asynccnt 0x0" ::: "memory");
        }
        __syncthreads();

        if (k0 <= q0w + 15) {                // wave-uniform causal skip
            // ---- S = Q K^T (two 16-key n-tiles, two K-chunks over d) ----
            v8f s0 = {}, s1 = {};
#pragma unroll
            for (int c = 0; c < 2; ++c) {
                v16bf kb0 = load_frag(&Ks[(0  + r) * KS_STRIDE + c * 32 + half * 16]);
                v16bf kb1 = load_frag(&Ks[(16 + r) * KS_STRIDE + c * 32 + half * 16]);
                s0 = wmma_bf16(qf[c], kb0, s0);
                s1 = wmma_bf16(qf[c], kb1, s1);
            }

            // ---- scale + causal mask + online softmax ----
            float p0[8], p1[8], alpha[8];
#pragma unroll
            for (int v = 0; v < 8; ++v) {
                const int q = q0w + half * 8 + v;
                float a0 = s0[v] * 0.125f;                   // 1/sqrt(64)
                float a1 = s1[v] * 0.125f;
                if (k0 + r      > q) a0 = -1e30f;
                if (k0 + 16 + r > q) a1 = -1e30f;
                float mx = rmax16(fmaxf(a0, a1));
                const float mn = fmaxf(m[v], mx);
                alpha[v] = __expf(m[v] - mn);
                m[v] = mn;
                a0 = __expf(a0 - mn);
                a1 = __expf(a1 - mn);
                const float ps = rsum16(a0 + a1);
                l[v] = l[v] * alpha[v] + ps;
                p0[v] = a0; p1[v] = a1;
            }
#pragma unroll
            for (int j = 0; j < 4; ++j)
#pragma unroll
                for (int v = 0; v < 8; ++v) yacc[j][v] *= alpha[v];

            // ---- C-layout -> A-layout for P via per-wave LDS ----
#pragma unroll
            for (int v = 0; v < 8; ++v) {
                const int row = half * 8 + v;
                Pw[row * 32 + r]      = f2bf(p0[v]);
                Pw[row * 32 + 16 + r] = f2bf(p1[v]);
            }
            // same-wave LDS RAW: DS ops are in-order within a wave
            v16bf pf = load_frag(&Pw[r * 32 + half * 16]);

            // ---- y += P (16x32) * V (32x64), four 16-col d-tiles ----
#pragma unroll
            for (int j = 0; j < 4; ++j) {
                v16bf vb = load_frag(&Vt[(j * 16 + r) * VT_STRIDE + half * 16]);
                yacc[j] = wmma_bf16(pf, vb, yacc[j]);
            }
        }
    }

    // ---- normalize and store y as bf16, [B,T,C] head-major ----
    float rl[8];
#pragma unroll
    for (int v = 0; v < 8; ++v) rl[v] = 1.0f / l[v];
#pragma unroll
    for (int j = 0; j < 4; ++j)
#pragma unroll
        for (int v = 0; v < 8; ++v) {
            const int q = q0w + half * 8 + v;
            ybf[(size_t)(b * T + q) * 1024 + h * 64 + j * 16 + r] =
                f2bf(yacc[j][v] * rl[v]);
        }
}

// ---------------------------------------------------------------------------
// launch
// ---------------------------------------------------------------------------
extern "C" void kernel_launch(void* const* d_in, const int* in_sizes, int n_in,
                              void* d_out, int out_size, void* d_ws, size_t ws_size,
                              hipStream_t stream) {
    constexpr int B = 32, T = 512, C = 1024;
    constexpr int M = B * T;                 // 16384

    const float* x    = (const float*)d_in[0];
    const float* Wqkv = (const float*)d_in[1];
    const float* bqkv = (const float*)d_in[2];
    const float* Wout = (const float*)d_in[3];
    const float* bout = (const float*)d_in[4];
    float* out = (float*)d_out;

    u16* x_bf    = (u16*)d_ws;                             // M*C
    u16* wqkv_bf = x_bf    + (size_t)M * C;                // 3C*C
    u16* wout_bf = wqkv_bf + (size_t)3 * C * C;            // C*C
    u16* qkv_bf  = wout_bf + (size_t)C * C;                // M*3C
    u16* y_bf    = qkv_bf  + (size_t)M * 3 * C;            // M*C

    f32_to_bf16_kernel<<<4096, 256, 0, stream>>>(x,    x_bf,    M * C);
    f32_to_bf16_kernel<<<2048, 256, 0, stream>>>(Wqkv, wqkv_bf, 3 * C * C);
    f32_to_bf16_kernel<<<1024, 256, 0, stream>>>(Wout, wout_bf, C * C);

    // qkv = x @ Wqkv^T + bqkv   -> bf16 [M, 3C]
    gemm_bf16_wmma<true><<<dim3(M / 128, (3 * C) / 64), 128, 0, stream>>>(
        x_bf, wqkv_bf, bqkv, qkv_bf, M, 3 * C, C);

    // flash attention -> bf16 y [M, C]
    attn_flash_wmma<<<dim3(T / 128, 16, B), 256, 0, stream>>>(qkv_bf, y_bf);

    // out = y @ Wout^T + bout  -> fp32 [M, C]
    gemm_bf16_wmma<false><<<dim3(M / 128, C / 64), 128, 0, stream>>>(
        y_bf, wout_bf, bout, out, M, C, C);
}
